// mPLUG_69922067579412
// MI455X (gfx1250) — matrix operand order repaired and checked by reading, not compiled, not run on hardware
//
#include <hip/hip_runtime.h>
#include <hip/hip_bf16.h>

typedef __attribute__((ext_vector_type(16))) __bf16 v16bf;
typedef __attribute__((ext_vector_type(2)))  __bf16 v2bf;
typedef __attribute__((ext_vector_type(2)))  float  v2f;
typedef __attribute__((ext_vector_type(8)))  float  v8f;

union FragBF { v16bf v; unsigned short u[16]; unsigned int d[8]; };

__device__ __forceinline__ unsigned int f2bf_pk(float lo, float hi) {
    v2f f = {lo, hi};
    v2bf p = __builtin_convertvector(f, v2bf);   // v_cvt_pk_bf16_f32
    unsigned int r;
    __builtin_memcpy(&r, &p, 4);
    return r;
}

// Registers holding one K-step's staging data (loaded early, stored late).
struct StageRegs { float a0[8], a1[8], b0[4], b1[4]; };

static __device__ __forceinline__ void stage_load(
    const float* __restrict__ A, int lda,
    const float* __restrict__ B, int ldb,
    int M, int N, int m0, int n0, int k0, int tid, StageRegs& sr)
{
    // A tile 128x32: 8 packed pairs per thread (pairs merge into global_load_b64)
    #pragma unroll
    for (int it = 0; it < 8; ++it) {
        int pidx = tid + it * 256;               // 0..2047
        int r = pidx >> 4, kp = pidx & 15;
        int gr = m0 + r; gr = gr < M ? gr : (M - 1);
        const float* ap = &A[(long)gr * lda + (k0 + 2 * kp)];
        sr.a0[it] = ap[0];
        sr.a1[it] = ap[1];
    }
    // B tile 32x64: 4 pairs per thread (two rows each, lane-coalesced)
    #pragma unroll
    for (int it = 0; it < 4; ++it) {
        int pidx = tid + it * 256;               // 0..1023
        int n = pidx & 63, kp = pidx >> 6;
        int gc = n0 + n; gc = gc < N ? gc : (N - 1);
        sr.b0[it] = B[(long)(k0 + 2 * kp) * ldb + gc];
        sr.b1[it] = B[(long)(k0 + 2 * kp + 1) * ldb + gc];
    }
}

static __device__ __forceinline__ void stage_store(
    unsigned int* __restrict__ sA, unsigned int* __restrict__ sBt,
    int tid, const StageRegs& sr)
{
    #pragma unroll
    for (int it = 0; it < 8; ++it) {
        int pidx = tid + it * 256;
        int r = pidx >> 4, kp = pidx & 15;
        sA[r * 20 + kp] = f2bf_pk(sr.a0[it], sr.a1[it]);
    }
    #pragma unroll
    for (int it = 0; it < 4; ++it) {
        int pidx = tid + it * 256;
        int n = pidx & 63, kp = pidx >> 6;
        sBt[n * 20 + kp] = f2bf_pk(sr.b0[it], sr.b1[it]);
    }
}

// ---------------------------------------------------------------------------
// Tiled bf16-WMMA GEMM: C[M,N] = act(A[M,K] @ B[K,N] + bias)
// Block tile 128x64, K-step 32, DOUBLE-BUFFERED LDS. 8 waves = 4(M) x 2(N);
// each wave owns a 32x32 sub-tile = 2x2 WMMA tiles (4 WMMAs per K-step).
// Per iteration: 1 barrier; frag ds_load_b128 (cur buf); issue next-tile
// global loads; 4x v_wmma; cvt + ds_store into other buf.
// ---------------------------------------------------------------------------
__global__ __launch_bounds__(256) void gemm_bf16_wmma(
    const float* __restrict__ A, int lda,
    const float* __restrict__ B, int ldb,
    float* __restrict__ C, int ldc,
    const float* __restrict__ bias,
    int M, int N, int K, int do_relu)
{
    __shared__ __align__(16) unsigned int sA[2][128 * 20];  // pad stride 20 dwords
    __shared__ __align__(16) unsigned int sBt[2][64 * 20];  // B transposed [col][kpair]

    const int tid  = threadIdx.x;
    const int lane = tid & 31;
    const int wave = tid >> 5;
    const int wm   = wave & 3;          // M group: rows wm*32 .. +32
    const int wn   = wave >> 2;         // N group: cols wn*32 .. +32
    const int m0   = blockIdx.y * 128;
    const int n0   = blockIdx.x * 64;
    const int half = lane >> 4;         // ISA 7.12.2 lane-half
    const int l16  = lane & 15;

    v8f acc00 = {}, acc01 = {}, acc10 = {}, acc11 = {};

    StageRegs sr;
    stage_load(A, lda, B, ldb, M, N, m0, n0, 0, tid, sr);
    stage_store(sA[0], sBt[0], tid, sr);

    const int nsteps = K >> 5;
    for (int t = 0; t < nsteps; ++t) {
        __syncthreads();
        const unsigned int* cA = sA[t & 1];
        const unsigned int* cB = sBt[t & 1];

        // A fragments: lane(half,m): K = half*8+0..7 and 16+half*8+0..7
        FragBF a0, a1;
        {
            int r0 = (wm * 32 + l16) * 20 + half * 4;
            __builtin_memcpy(&a0.d[0], &cA[r0],     16);
            __builtin_memcpy(&a0.d[4], &cA[r0 + 8], 16);
            int r1 = (wm * 32 + 16 + l16) * 20 + half * 4;
            __builtin_memcpy(&a1.d[0], &cA[r1],     16);
            __builtin_memcpy(&a1.d[4], &cA[r1 + 8], 16);
        }
        // B fragments: lane(half,n): K = half*8+0..7 and 16+half*8+0..7
        FragBF b0, b1;
        {
            int c0 = (wn * 32 + l16) * 20 + half * 4;
            __builtin_memcpy(&b0.d[0], &cB[c0],     16);
            __builtin_memcpy(&b0.d[4], &cB[c0 + 8], 16);
            int c1 = (wn * 32 + 16 + l16) * 20 + half * 4;
            __builtin_memcpy(&b1.d[0], &cB[c1],     16);
            __builtin_memcpy(&b1.d[4], &cB[c1 + 8], 16);
        }

        // Issue next tile's global loads now; WMMAs below only wait on DScnt,
        // so the load latency overlaps matrix math.
        const bool have_next = (t + 1) < nsteps;
        if (have_next)
            stage_load(A, lda, B, ldb, M, N, m0, n0, (t + 1) * 32, tid, sr);

        acc00 = __builtin_amdgcn_wmma_f32_16x16x32_bf16(false, a0.v, false, b0.v,
                                                        (short)0, acc00, false, false);
        acc01 = __builtin_amdgcn_wmma_f32_16x16x32_bf16(false, a0.v, false, b1.v,
                                                        (short)0, acc01, false, false);
        acc10 = __builtin_amdgcn_wmma_f32_16x16x32_bf16(false, a1.v, false, b0.v,
                                                        (short)0, acc10, false, false);
        acc11 = __builtin_amdgcn_wmma_f32_16x16x32_bf16(false, a1.v, false, b1.v,
                                                        (short)0, acc11, false, false);

        if (have_next)
            stage_store(sA[(t + 1) & 1], sBt[(t + 1) & 1], tid, sr);
    }

    // C/D layout: lane(half,n), VGPR v -> (M = half*8 + v, N = n)
    const int c0 = n0 + wn * 32 + l16;
    const int c1 = c0 + 16;
    const bool okc0 = c0 < N;
    const bool okc1 = c1 < N;
    const float bv0 = (bias && okc0) ? bias[c0] : 0.f;
    const float bv1 = (bias && okc1) ? bias[c1] : 0.f;

    #pragma unroll
    for (int mt = 0; mt < 2; ++mt) {
        #pragma unroll
        for (int v = 0; v < 8; ++v) {
            int row = m0 + wm * 32 + mt * 16 + half * 8 + v;
            if (row < M) {
                float f0 = ((mt == 0) ? acc00[v] : acc10[v]) + bv0;
                float f1 = ((mt == 0) ? acc01[v] : acc11[v]) + bv1;
                if (do_relu) { f0 = fmaxf(f0, 0.f); f1 = fmaxf(f1, 0.f); }
                if (okc0) C[(long)row * ldc + c0] = f0;
                if (okc1) C[(long)row * ldc + c1] = f1;
            }
        }
    }
}

// ---------------------------------------------------------------------------
// IOU epilogue: out[b,h,i,j] = relu(Ps[b,i]+Pc[b,(i+j)>>1]+Pe[b,j]+b1) . W2[:,h] + b2[h]
// ---------------------------------------------------------------------------
__global__ __launch_bounds__(256) void iou_epilogue(
    const float* __restrict__ Ps, const float* __restrict__ Pc,
    const float* __restrict__ Pe, const float* __restrict__ b1,
    const float* __restrict__ W2, const float* __restrict__ b2,
    float* __restrict__ out)
{
    const int jt = blockIdx.x, it = blockIdx.y, b = blockIdx.z;
    const int i0 = it * 16, j0 = jt * 16;
    const int c0 = (i0 + j0) >> 1;

    __shared__ float ss[16][257];
    __shared__ float cc[16][257];
    __shared__ float ee[16][257];
    __shared__ float wch[256 * 3];
    __shared__ float bch[256];

    const int tid = threadIdx.x;
    const int ii = tid >> 4, jj = tid & 15;
    const int gi = i0 + ii, gj = j0 + jj;
    const int ci = ((gi + gj) >> 1) - c0;    // 0..15 within tile

    float a0 = 0.f, a1 = 0.f, a2 = 0.f;

    for (int d0 = 0; d0 < 768; d0 += 256) {
        __syncthreads();
        #pragma unroll
        for (int t = 0; t < 16; ++t) {
            int idx = tid + t * 256;          // 0..4095
            int r = idx >> 8, c = idx & 255;
            int rs = min(i0 + r, 128);
            int rc = min(c0 + r, 128);
            int re = min(j0 + r, 128);
            ss[r][c] = Ps[((long)b * 129 + rs) * 768 + d0 + c];
            cc[r][c] = Pc[((long)b * 129 + rc) * 768 + d0 + c];
            ee[r][c] = Pe[((long)b * 129 + re) * 768 + d0 + c];
        }
        bch[tid] = b1[d0 + tid];
        #pragma unroll
        for (int t = 0; t < 3; ++t) {
            int idx = tid + t * 256;          // 0..767 into W2[d0*3 ..]
            wch[idx] = W2[(long)d0 * 3 + idx];
        }
        __syncthreads();

        for (int d = 0; d < 256; ++d) {
            float v = ss[ii][d] + cc[ci][d] + ee[jj][d] + bch[d];
            float r = fmaxf(v, 0.f);
            a0 = fmaf(r, wch[d * 3 + 0], a0);
            a1 = fmaf(r, wch[d * 3 + 1], a1);
            a2 = fmaf(r, wch[d * 3 + 2], a2);
        }
    }

    if (gi < 129 && gj < 129) {
        out[(((long)b * 3 + 0) * 129 + gi) * 129 + gj] = a0 + b2[0];
        out[(((long)b * 3 + 1) * 129 + gi) * 129 + gj] = a1 + b2[1];
        out[(((long)b * 3 + 2) * 129 + gi) * 129 + gj] = a2 + b2[2];
    }
}

// ---------------------------------------------------------------------------
// Visual logits: out[row,head] = H_head[row,:] . w2_head + b2_head  (wave per dot)
// ---------------------------------------------------------------------------
__global__ __launch_bounds__(256) void visual_logits(
    const float* __restrict__ Hs, const float* __restrict__ He,
    const float* __restrict__ Hc,
    const float* __restrict__ ws2, const float* __restrict__ we2,
    const float* __restrict__ wc2,
    const float* __restrict__ bs2, const float* __restrict__ be2,
    const float* __restrict__ bc2,
    float* __restrict__ out)
{
    int wv   = (blockIdx.x * 256 + threadIdx.x) >> 5;
    int lane = threadIdx.x & 31;
    int head = wv % 3;
    int row  = wv / 3;
    if (row >= 1032) return;                         // uniform per wave
    const float* H = (head == 0) ? Hs : (head == 1) ? He : Hc;
    const float* w = (head == 0) ? ws2 : (head == 1) ? we2 : wc2;
    float bb = (head == 0) ? bs2[0] : (head == 1) ? be2[0] : bc2[0];
    float acc = 0.f;
    for (int d = lane; d < 768; d += 32)
        acc = fmaf(H[(long)row * 768 + d], w[d], acc);
    #pragma unroll
    for (int off = 16; off; off >>= 1) acc += __shfl_xor(acc, off);
    if (lane == 0) out[row * 3 + head] = acc + bb;
}

// ---------------------------------------------------------------------------
// IOU mask map (closed-form membership test of the reference's 4 loops)
// ---------------------------------------------------------------------------
__global__ void mask_kernel(float* __restrict__ out) {
    int t = blockIdx.x * 256 + threadIdx.x;
    if (t >= 129 * 129) return;
    int i = t / 129, j = t % 129, d = j - i;
    bool m = false;
    if (d >= 1 && d <= 16) m = true;
    if (!(i & 1) && i <= 110 && d >= 18 && d <= 32 && !(d & 1)) m = true;
    if (!(i & 3) && i <= 92  && d >= 36 && d <= 64 && !(d & 3)) m = true;
    if (!(i & 7) && i <= 56  && d >= 72 && !(d & 7)) m = true;
    out[t] = m ? 1.f : 0.f;
}

// ---------------------------------------------------------------------------
static inline dim3 gemm_grid(int M, int N) { return dim3((N + 63) / 64, (M + 127) / 128); }

extern "C" void kernel_launch(void* const* d_in, const int* in_sizes, int n_in,
                              void* d_out, int out_size, void* d_ws, size_t ws_size,
                              hipStream_t stream) {
    (void)in_sizes; (void)n_in; (void)out_size; (void)ws_size;
    const float* text   = (const float*)d_in[0];   // [8,128,768]
    const float* obj    = (const float*)d_in[1];   // [8,129,768]
    const float* t_w1   = (const float*)d_in[2];
    const float* t_b1   = (const float*)d_in[3];
    const float* t_w2   = (const float*)d_in[4];   // [768,30522]
    const float* t_b2   = (const float*)d_in[5];
    const float* m2_w   = (const float*)d_in[6];   // [768,2304]
    const float* m2_b   = (const float*)d_in[7];
    const float* m3_w1  = (const float*)d_in[8];   // [2304,768]
    const float* m3_b1  = (const float*)d_in[9];
    const float* m3_w2  = (const float*)d_in[10];  // [768,3]
    const float* m3_b2  = (const float*)d_in[11];
    const float* ms_w1  = (const float*)d_in[12];
    const float* ms_b1  = (const float*)d_in[13];
    const float* ms_w2  = (const float*)d_in[14];
    const float* ms_b2  = (const float*)d_in[15];
    const float* me_w1  = (const float*)d_in[16];
    const float* me_b1  = (const float*)d_in[17];
    const float* me_w2  = (const float*)d_in[18];
    const float* me_b2  = (const float*)d_in[19];
    const float* mc_w1  = (const float*)d_in[20];
    const float* mc_b1  = (const float*)d_in[21];
    const float* mc_w2  = (const float*)d_in[22];
    const float* mc_b2  = (const float*)d_in[23];

    // output layout (flat, in return order)
    float* out_text = (float*)d_out;                    // 8*128*30522
    float* out_vis  = out_text + 31254528;              // 8*129*3
    float* out_iou  = out_vis + 3096;                   // 8*3*129*129
    float* out_mask = out_iou + 399384;                 // 129*129

    // workspace layout (floats)
    float* X1 = (float*)d_ws;                           // [1024,768]
    float* h  = X1 + 1024 * 768;                        // [1032,2304]
    float* Ps = h  + 1032 * 2304;                       // [1032,768]
    float* Pc = Ps + 1032 * 768;
    float* Pe = Pc + 1032 * 768;
    float* Hs = Pe + 1032 * 768;
    float* He = Hs + 1032 * 768;
    float* Hc = He + 1032 * 768;

    dim3 blk(256);

    // 1) text head hidden: X1 = relu(text @ t_w1 + t_b1)   [1024,768]
    gemm_bf16_wmma<<<gemm_grid(1024, 768), blk, 0, stream>>>(
        text, 768, t_w1, 768, X1, 768, t_b1, 1024, 768, 768, 1);
    // 2) text logits: out_text = X1 @ t_w2 + t_b2          [1024,30522]
    gemm_bf16_wmma<<<gemm_grid(1024, 30522), blk, 0, stream>>>(
        X1, 768, t_w2, 30522, out_text, 30522, t_b2, 1024, 30522, 768, 0);
    // 3) object projection: h = relu(obj @ m2_w + m2_b)    [1032,2304]
    gemm_bf16_wmma<<<gemm_grid(1032, 2304), blk, 0, stream>>>(
        obj, 768, m2_w, 2304, h, 2304, m2_b, 1032, 2304, 768, 1);
    // h column slices: hidden_s = h[:,0:768], hidden_e = h[:,768:1536], hidden_c = h[:,1536:2304]
    // 4-6) decomposed IOU partials (W1 row-blocks: s -> 0:768, c -> 768:1536, e -> 1536:2304)
    gemm_bf16_wmma<<<gemm_grid(1032, 768), blk, 0, stream>>>(
        h + 0,    2304, m3_w1 + 0,          768, Ps, 768, nullptr, 1032, 768, 768, 0);
    gemm_bf16_wmma<<<gemm_grid(1032, 768), blk, 0, stream>>>(
        h + 1536, 2304, m3_w1 + 768 * 768,  768, Pc, 768, nullptr, 1032, 768, 768, 0);
    gemm_bf16_wmma<<<gemm_grid(1032, 768), blk, 0, stream>>>(
        h + 768,  2304, m3_w1 + 1536 * 768, 768, Pe, 768, nullptr, 1032, 768, 768, 0);
    // 7-9) visual head hiddens
    gemm_bf16_wmma<<<gemm_grid(1032, 768), blk, 0, stream>>>(
        h + 0,    2304, ms_w1, 768, Hs, 768, ms_b1, 1032, 768, 768, 1);
    gemm_bf16_wmma<<<gemm_grid(1032, 768), blk, 0, stream>>>(
        h + 768,  2304, me_w1, 768, He, 768, me_b1, 1032, 768, 768, 1);
    gemm_bf16_wmma<<<gemm_grid(1032, 768), blk, 0, stream>>>(
        h + 1536, 2304, mc_w1, 768, Hc, 768, mc_b1, 1032, 768, 768, 1);
    // 10) fused IOU epilogue -> out_iou [8,3,129,129]
    iou_epilogue<<<dim3(9, 9, 8), blk, 0, stream>>>(
        Ps, Pc, Pe, m3_b1, m3_w2, m3_b2, out_iou);
    // 11) visual logits -> out_vis [1032,3]
    visual_logits<<<dim3((1032 * 3 * 32 + 255) / 256), blk, 0, stream>>>(
        Hs, He, Hc, ms_w2, me_w2, mc_w2, ms_b2, me_b2, mc_b2, out_vis);
    // 12) mask map -> out_mask [129,129]
    mask_kernel<<<dim3((129 * 129 + 255) / 256), blk, 0, stream>>>(out_mask);
}
